// Attention_3272765079862
// MI455X (gfx1250) — compile-verified, matrix operand order checked by reading, not run on hardware
//
#include <hip/hip_runtime.h>

#define DIMN 2048
#define HD   128
#define NHD  16
#define SEQL 2048
#define HALFD (HD / 2)

typedef __attribute__((ext_vector_type(16))) __bf16 v16bf;
typedef __attribute__((ext_vector_type(8)))  float  v8f;

union FragU { uint4 u[2]; v16bf v; };

__device__ __forceinline__ unsigned short f2bf(float f) {
  unsigned int u = __float_as_uint(f);
  unsigned int r = (u + 0x7FFFu + ((u >> 16) & 1u)) >> 16;
  return (unsigned short)r;
}

// Load a 16-bit A/B WMMA fragment: per lane, 8 elems at +0 and 8 at +16
// (ISA 16-bit 16x32 layout: VGPR0-3 = K[khalf..khalf+7], VGPR4-7 = K[khalf+16..+23]).
__device__ __forceinline__ v16bf load_frag_g(const unsigned short* p) {
  FragU f;
  f.u[0] = *(const uint4*)(p);
  f.u[1] = *(const uint4*)(p + 16);
  return f.v;
}

// ---------------------------------------------------------------- convert
__global__ void cvt_kernel(const float* __restrict__ in,
                           unsigned short* __restrict__ out, int n4) {
  int i = blockIdx.x * blockDim.x + threadIdx.x;
  if (i < n4) {
    float4 f = ((const float4*)in)[i];
    ushort4 o;
    o.x = f2bf(f.x); o.y = f2bf(f.y); o.z = f2bf(f.z); o.w = f2bf(f.w);
    ((ushort4*)out)[i] = o;
  }
}

// ---------------------------------------------------------------- QKV GEMM
// Block: 8 waves (2 row x 4 col), block tile 64 tokens x 256 cols (2 heads).
// Wave tile 32x64 = 2x4 C-fragments: 12 b128 loads per 8 WMMAs (1.5/WMMA).
// Epilogue via 64KB LDS tile fuses rmsnorm + rope for q/k, transposed V store.
__global__ __launch_bounds__(256) void gemm_qkv_kernel(
    const unsigned short* __restrict__ xb,   // [T, DIM] bf16
    const unsigned short* __restrict__ wb,   // [3*DIM, DIM] bf16
    const float* __restrict__ rope_cos,      // [T, 64]
    const float* __restrict__ rope_sin,      // [T, 64]
    const float* __restrict__ qnw,           // [128]
    const float* __restrict__ knw,           // [128]
    unsigned short* __restrict__ qb,         // [T, DIM] bf16 (normed+roped)
    unsigned short* __restrict__ kb,         // [T, DIM] bf16 (normed+roped)
    unsigned short* __restrict__ vt,         // [DIM, T] bf16 (transposed V)
    int Ttot) {
  __shared__ float tile[64][256];

  const int w    = threadIdx.x >> 5;
  const int lane = threadIdx.x & 31;
  const int t0   = blockIdx.y * 64;
  const int j0   = blockIdx.x * 256;
  const int rb   = (w & 1) * 32;   // wave row offset
  const int cb   = (w >> 1) * 64;  // wave col offset
  const int rl   = lane & 15;
  const int kh   = (lane >> 4) << 3;

  const unsigned short* ap = xb + (size_t)(t0 + rb + rl) * DIMN + kh;
  const unsigned short* bp = wb + (size_t)(j0 + cb + rl) * DIMN + kh;

  v8f cacc[2][4] = {};
#pragma unroll 2
  for (int k = 0; k < DIMN; k += 32) {
    __builtin_prefetch(ap + k + 1024, 0, 1);
    __builtin_prefetch(bp + k + 1024, 0, 1);
    v16bf a0 = load_frag_g(ap + k);
    v16bf a1 = load_frag_g(ap + (size_t)16 * DIMN + k);
#pragma unroll
    for (int ci = 0; ci < 4; ++ci) {
      v16bf b = load_frag_g(bp + (size_t)(ci * 16) * DIMN + k);
      cacc[0][ci] = __builtin_amdgcn_wmma_f32_16x16x32_bf16(
          false, a0, false, b, (short)0, cacc[0][ci], false, false);
      cacc[1][ci] = __builtin_amdgcn_wmma_f32_16x16x32_bf16(
          false, a1, false, b, (short)0, cacc[1][ci], false, false);
    }
  }

  const int mo = (lane >> 4) << 3;
#pragma unroll
  for (int ri = 0; ri < 2; ++ri)
#pragma unroll
    for (int ci = 0; ci < 4; ++ci)
#pragma unroll
      for (int v = 0; v < 8; ++v)
        tile[rb + ri * 16 + mo + v][cb + ci * 16 + rl] = cacc[ri][ci][v];
  __syncthreads();

  const int which = j0 / DIMN;            // 0=q 1=k 2=v (block-uniform)
  const int jm    = j0 % DIMN;            // col offset inside that matrix

#pragma unroll
  for (int it = 0; it < 8; ++it) {
    const int p     = it * 256 + threadIdx.x;
    const int r     = p >> 5;             // 0..63
    const int c     = (p & 31) << 3;      // 0..248
    const int token = t0 + r;
    float vals[8];
#pragma unroll
    for (int i = 0; i < 8; ++i) vals[i] = tile[r][c + i];

    if (which == 2) {
#pragma unroll
      for (int i = 0; i < 8; ++i)
        vt[(size_t)(jm + c + i) * Ttot + token] = f2bf(vals[i]);
    } else {
      float ss = 0.f;
#pragma unroll
      for (int i = 0; i < 8; ++i) ss += vals[i] * vals[i];
      // threads of one (row, head) occupy lanes 0-15 or 16-31 of a wave
#pragma unroll
      for (int m = 1; m < 16; m <<= 1) ss += __shfl_xor(ss, m, 32);
      float rstd = rsqrtf(ss * (1.0f / 128.0f) + 1e-5f);
      const float* nw = (which == 0) ? qnw : knw;
      const int d = c & 127;              // position within head
      union { unsigned short s[8]; uint4 u; } ov;
#pragma unroll
      for (int i = 0; i < 4; ++i) {
        int dd   = d + 2 * i;
        float e  = vals[2 * i]     * rstd * nw[dd];
        float o  = vals[2 * i + 1] * rstd * nw[dd + 1];
        float cs = rope_cos[(size_t)token * HALFD + (dd >> 1)];
        float sn = rope_sin[(size_t)token * HALFD + (dd >> 1)];
        ov.s[2 * i]     = f2bf(e * cs - o * sn);
        ov.s[2 * i + 1] = f2bf(e * sn + o * cs);
      }
      unsigned short* dst =
          ((which == 0) ? qb : kb) + (size_t)token * DIMN + jm + c;
      *(uint4*)dst = ov.u;
    }
  }
}

// ---------------------------------------------------------------- attention
// Block = 8 waves, each wave owns one 16-token q-tile of one (b,h).
// Flash-style online softmax; 32 keys per iteration; 16 WMMA / 32 b128 loads.
__global__ __launch_bounds__(256) void attn_kernel(
    const unsigned short* __restrict__ qb,
    const unsigned short* __restrict__ kb,
    const unsigned short* __restrict__ vt,
    unsigned short* __restrict__ att,       // [T, DIM] bf16
    int Ttot) {
  __shared__ __align__(16) unsigned short ldsP[8][16][32];

  const int w    = threadIdx.x >> 5;
  const int lane = threadIdx.x & 31;
  const int nqc  = SEQL / 128;
  const int qc   = blockIdx.x % nqc;
  const int tmp  = blockIdx.x / nqc;
  const int h    = tmp % NHD;
  const int b    = tmp / NHD;
  const int base = b * SEQL;
  const int t0   = base + qc * 128 + (w << 4);
  const int rl   = lane & 15;
  const int kh   = (lane >> 4) << 3;
  const float scale = 0.08838834764831845f;  // 1/sqrt(128)

  v16bf qf[4];
  {
    const unsigned short* qp = qb + (size_t)(t0 + rl) * DIMN + h * HD + kh;
#pragma unroll
    for (int kc = 0; kc < 4; ++kc) qf[kc] = load_frag_g(qp + kc * 32);
  }

  v8f acc[8] = {};
  float mrow[8], lrow[8];
#pragma unroll
  for (int v = 0; v < 8; ++v) { mrow[v] = -1e30f; lrow[v] = 0.f; }

  const int mo = (lane >> 4) << 3;

  for (int j = 0; j < SEQL; j += 32) {
    // S = Q K^T for 32 keys (two 16x16 C-fragments)
    v8f s0 = {}, s1 = {};
    const unsigned short* kp0 =
        kb + (size_t)(base + j + rl) * DIMN + h * HD + kh;
    const unsigned short* kp1 = kp0 + (size_t)16 * DIMN;
#pragma unroll
    for (int kc = 0; kc < 4; ++kc) {
      s0 = __builtin_amdgcn_wmma_f32_16x16x32_bf16(
          false, qf[kc], false, load_frag_g(kp0 + kc * 32), (short)0, s0,
          false, false);
      s1 = __builtin_amdgcn_wmma_f32_16x16x32_bf16(
          false, qf[kc], false, load_frag_g(kp1 + kc * 32), (short)0, s1,
          false, false);
    }

    // online softmax over the 32 new keys (row = v + mo, cols across lanes)
    float p0[8], p1[8], alpha[8];
#pragma unroll
    for (int v = 0; v < 8; ++v) {
      p0[v] = s0[v] * scale;
      p1[v] = s1[v] * scale;
      float mx = fmaxf(p0[v], p1[v]);
#pragma unroll
      for (int msk = 1; msk < 16; msk <<= 1)
        mx = fmaxf(mx, __shfl_xor(mx, msk, 32));
      float mn = fmaxf(mrow[v], mx);
      alpha[v] = __expf(mrow[v] - mn);
      mrow[v]  = mn;
      p0[v] = __expf(p0[v] - mn);
      p1[v] = __expf(p1[v] - mn);
      float rs = p0[v] + p1[v];
#pragma unroll
      for (int msk = 1; msk < 16; msk <<= 1) rs += __shfl_xor(rs, msk, 32);
      lrow[v] = lrow[v] * alpha[v] + rs;
    }
#pragma unroll
    for (int dt = 0; dt < 8; ++dt)
#pragma unroll
      for (int v = 0; v < 8; ++v) acc[dt][v] *= alpha[v];

    // transpose P (C-frag) -> A-frag through per-wave LDS slice
#pragma unroll
    for (int v = 0; v < 8; ++v) {
      ldsP[w][v + mo][rl]      = f2bf(p0[v]);
      ldsP[w][v + mo][rl + 16] = f2bf(p1[v]);
    }
    v16bf pf;
    {
      FragU f;
      f.u[0] = *(const uint4*)&ldsP[w][rl][kh];
      f.u[1] = *(const uint4*)&ldsP[w][rl][kh + 16];
      pf = f.v;
    }

    // O += P V : Vt[d][t] gives contiguous-in-key B fragments
#pragma unroll
    for (int dt = 0; dt < 8; ++dt) {
      const unsigned short* vp =
          vt + (size_t)(h * HD + dt * 16 + rl) * Ttot + base + j + kh;
      FragU f;
      f.u[0] = *(const uint4*)vp;
      f.u[1] = *(const uint4*)(vp + 16);
      acc[dt] = __builtin_amdgcn_wmma_f32_16x16x32_bf16(
          false, pf, false, f.v, (short)0, acc[dt], false, false);
    }
  }

  // normalize and store att (bf16, [T, DIM] with head offset)
#pragma unroll
  for (int dt = 0; dt < 8; ++dt) {
#pragma unroll
    for (int v = 0; v < 8; ++v) {
      int token = t0 + v + mo;
      att[(size_t)token * DIMN + h * HD + dt * 16 + rl] =
          f2bf(acc[dt][v] / lrow[v]);
    }
  }
}

// ---------------------------------------------------------------- out GEMM
// Same 64x256 block / 32x64 wave tiling as gemm_qkv; f32 output.
__global__ __launch_bounds__(256) void gemm_out_kernel(
    const unsigned short* __restrict__ ab,   // [T, DIM] bf16 (att)
    const unsigned short* __restrict__ wb,   // [DIM, DIM] bf16 (wo)
    float* __restrict__ out,                 // [T, DIM] f32
    int Ttot) {
  __shared__ float tile[64][256];

  const int w    = threadIdx.x >> 5;
  const int lane = threadIdx.x & 31;
  const int t0   = blockIdx.y * 64;
  const int j0   = blockIdx.x * 256;
  const int rb   = (w & 1) * 32;
  const int cb   = (w >> 1) * 64;
  const int rl   = lane & 15;
  const int kh   = (lane >> 4) << 3;

  const unsigned short* ap = ab + (size_t)(t0 + rb + rl) * DIMN + kh;
  const unsigned short* bp = wb + (size_t)(j0 + cb + rl) * DIMN + kh;

  v8f cacc[2][4] = {};
#pragma unroll 2
  for (int k = 0; k < DIMN; k += 32) {
    __builtin_prefetch(ap + k + 1024, 0, 1);
    __builtin_prefetch(bp + k + 1024, 0, 1);
    v16bf a0 = load_frag_g(ap + k);
    v16bf a1 = load_frag_g(ap + (size_t)16 * DIMN + k);
#pragma unroll
    for (int ci = 0; ci < 4; ++ci) {
      v16bf b = load_frag_g(bp + (size_t)(ci * 16) * DIMN + k);
      cacc[0][ci] = __builtin_amdgcn_wmma_f32_16x16x32_bf16(
          false, a0, false, b, (short)0, cacc[0][ci], false, false);
      cacc[1][ci] = __builtin_amdgcn_wmma_f32_16x16x32_bf16(
          false, a1, false, b, (short)0, cacc[1][ci], false, false);
    }
  }

  const int mo = (lane >> 4) << 3;
#pragma unroll
  for (int ri = 0; ri < 2; ++ri)
#pragma unroll
    for (int ci = 0; ci < 4; ++ci)
#pragma unroll
      for (int v = 0; v < 8; ++v)
        tile[rb + ri * 16 + mo + v][cb + ci * 16 + rl] = cacc[ri][ci][v];
  __syncthreads();

#pragma unroll
  for (int it = 0; it < 8; ++it) {
    const int p = it * 256 + threadIdx.x;
    const int r = p >> 5;
    const int c = (p & 31) << 3;
    float* dst = out + (size_t)(t0 + r) * DIMN + j0 + c;
    float4 f0 = make_float4(tile[r][c + 0], tile[r][c + 1],
                            tile[r][c + 2], tile[r][c + 3]);
    float4 f1 = make_float4(tile[r][c + 4], tile[r][c + 5],
                            tile[r][c + 6], tile[r][c + 7]);
    ((float4*)dst)[0] = f0;
    ((float4*)dst)[1] = f1;
  }
}

// ---------------------------------------------------------------- launch
extern "C" void kernel_launch(void* const* d_in, const int* in_sizes, int n_in,
                              void* d_out, int out_size, void* d_ws,
                              size_t ws_size, hipStream_t stream) {
  const float* x    = (const float*)d_in[0];
  const float* rc   = (const float*)d_in[1];
  const float* rs   = (const float*)d_in[2];
  // d_in[3] = cu, d_in[4] = max_length: uniform-length layout assumed (B*L)
  const float* wqkv = (const float*)d_in[5];
  const float* wo   = (const float*)d_in[6];
  const float* qnw  = (const float*)d_in[7];
  const float* knw  = (const float*)d_in[8];
  float* out = (float*)d_out;

  const int Ttot = in_sizes[0] / DIMN;      // 8192
  const int Bn   = Ttot / SEQL;             // 4

  char* ws = (char*)d_ws;
  size_t off = 0;
  auto take = [&](size_t bytes) {
    char* p = ws + off;
    off += (bytes + 255) & ~(size_t)255;
    return p;
  };
  unsigned short* xb   = (unsigned short*)take((size_t)Ttot * DIMN * 2);
  unsigned short* wqb  = (unsigned short*)take((size_t)3 * DIMN * DIMN * 2);
  unsigned short* wob  = (unsigned short*)take((size_t)DIMN * DIMN * 2);
  unsigned short* qbuf = (unsigned short*)take((size_t)Ttot * DIMN * 2);
  unsigned short* kbuf = (unsigned short*)take((size_t)Ttot * DIMN * 2);
  unsigned short* vtb  = (unsigned short*)take((size_t)Ttot * DIMN * 2);
  unsigned short* attb = (unsigned short*)take((size_t)Ttot * DIMN * 2);

  int n4;
  n4 = Ttot * DIMN / 4;
  cvt_kernel<<<(n4 + 255) / 256, 256, 0, stream>>>(x, xb, n4);
  n4 = 3 * DIMN * DIMN / 4;
  cvt_kernel<<<(n4 + 255) / 256, 256, 0, stream>>>(wqkv, wqb, n4);
  n4 = DIMN * DIMN / 4;
  cvt_kernel<<<(n4 + 255) / 256, 256, 0, stream>>>(wo, wob, n4);

  dim3 g1(3 * DIMN / 256, Ttot / 64);
  gemm_qkv_kernel<<<g1, 256, 0, stream>>>(xb, wqb, rc, rs, qnw, knw, qbuf,
                                          kbuf, vtb, Ttot);

  attn_kernel<<<Bn * NHD * (SEQL / 128), 256, 0, stream>>>(qbuf, kbuf, vtb,
                                                           attb, Ttot);

  dim3 g2(DIMN / 256, Ttot / 64);
  gemm_out_kernel<<<g2, 256, 0, stream>>>(attb, wob, out, Ttot);
}
